// MultiheadAttention_40561671143927
// MI455X (gfx1250) — matrix operand order repaired and checked by reading, not compiled, not run on hardware
//
#include <hip/hip_runtime.h>
#include <cstdint>

#define B_ 2
#define S_ 2048
#define D_ 2048
#define H_ 16
#define DH_ 128
#define M_ (B_*S_)        // 4096 rows (b*s)
#define N_QKV (3*D_)      // 6144
#define CLIP_ 8.0f
#define SMAX_SCALE 0.08838834764831845f   // 1/sqrt(128)

typedef __attribute__((ext_vector_type(16))) __bf16    v16bf;
typedef __attribute__((ext_vector_type(8)))  float     v8f;
typedef __attribute__((ext_vector_type(4)))  uint32_t  u32x4;
typedef __attribute__((ext_vector_type(4)))  int       v4i_t;

union Frag16 { v16bf v; u32x4 q[2]; };

__device__ __forceinline__ __bf16 f2bf(float f) {
  uint32_t u = __builtin_bit_cast(uint32_t, f);
  u += 0x7FFFu + ((u >> 16) & 1u);             // round-to-nearest-even
  uint16_t h = (uint16_t)(u >> 16);
  return __builtin_bit_cast(__bf16, h);
}
__device__ __forceinline__ float bf2f(__bf16 h) {
  uint32_t u = ((uint32_t)__builtin_bit_cast(uint16_t, h)) << 16;
  return __builtin_bit_cast(float, u);
}

// ---------------- async global->LDS staging (CDNA5 path w/ safe fallback) ---
#if __has_builtin(__builtin_amdgcn_global_load_async_to_lds_b128)
#define HAVE_ASYNC_CP 1
typedef __attribute__((address_space(1))) v4i_t as1_v4i;   // global src
typedef __attribute__((address_space(3))) v4i_t as3_v4i;   // LDS dst
#endif

// copy 8 bf16 (16B) from global to LDS
__device__ __forceinline__ void cp16_lds(__bf16* dst, const __bf16* src) {
#ifdef HAVE_ASYNC_CP
  __builtin_amdgcn_global_load_async_to_lds_b128(
      (as1_v4i*)src, (as3_v4i*)dst, 0, 0);
#else
  *(u32x4*)dst = *(const u32x4*)src;
#endif
}

__device__ __forceinline__ void wait_async0() {
#ifdef HAVE_ASYNC_CP
#if __has_builtin(__builtin_amdgcn_s_wait_asynccnt)
  __builtin_amdgcn_s_wait_asynccnt(0);
#else
  asm volatile("s_wait_asynccnt 0" ::: "memory");
#endif
#endif
}

// ---------------------------------------------------------------- convert ---
__global__ void cvt_f32_bf16(const float* __restrict__ in,
                             __bf16* __restrict__ out, int n) {
  for (int i = blockIdx.x * blockDim.x + threadIdx.x; i < n;
       i += gridDim.x * blockDim.x)
    out[i] = f2bf(in[i]);
}

// ------------------------------------------------------------------- GEMM ---
// C[M,N] = A[M,K] * BT[N,K]^T (+bias). Block tile 256x128, 8 waves,
// each wave: 32 M-rows x 128 N. K-step 32, double-buffered LDS with
// async global->LDS prefetch.
__global__ __launch_bounds__(256) void gemm_bf16_wmma(
    const __bf16* __restrict__ A,    // M x K row-major (bf16)
    const __bf16* __restrict__ BT,   // N x K row-major (= B transposed)
    const float*  __restrict__ bias, // length N
    __bf16* __restrict__ outB,       // used when clipToBf16 == 1
    float*  __restrict__ outF,       // used when clipToBf16 == 0
    int Mdim, int Ndim, int Kdim, int clipToBf16)
{
  __shared__ __align__(16) __bf16 sA[2][256][40];   // 32 data cols + pad
  __shared__ __align__(16) __bf16 sB[2][128][40];
  const int tid  = threadIdx.x;
  const int wave = tid >> 5;
  const int lane = tid & 31;
  const int lh   = lane >> 4;     // half-wave select
  const int ll   = lane & 15;
  const int m0 = blockIdx.x * 256;
  const int n0 = blockIdx.y * 128;

  v8f acc[2][8] = {};

  // staging: thread t owns A row (m0+t), 32 cols; B row (n0 + t/2), 16 cols
  const __bf16* Abase = A  + (size_t)(m0 + tid) * Kdim;
  const __bf16* Bbase = BT + (size_t)(n0 + (tid >> 1)) * Kdim + (tid & 1) * 16;
  const int brow_st = tid >> 1;
  const int bcol_st = (tid & 1) * 16;

  auto stage = [&](int buf, int k0) {
#pragma unroll
    for (int c = 0; c < 4; ++c)
      cp16_lds(&sA[buf][tid][c * 8], Abase + k0 + c * 8);
#pragma unroll
    for (int c = 0; c < 2; ++c)
      cp16_lds(&sB[buf][brow_st][bcol_st + c * 8], Bbase + k0 + c * 8);
  };

  stage(0, 0);
  wait_async0();
  __syncthreads();

  const int ar = wave * 32 + ll;
  const int ak = lh * 8;          // A frag: lanes 0-15 K0..7,K16..23
  const int bk = lh * 8;          // B frag: lanes 0-15 K0..7,K16..23

  for (int k0 = 0; k0 < Kdim; k0 += 32) {
    const int cur = (k0 >> 5) & 1;
    if (k0 + 32 < Kdim) stage(cur ^ 1, k0 + 32);

    Frag16 af0, af1;
    af0.q[0] = *(const u32x4*)&sA[cur][ar][ak];
    af0.q[1] = *(const u32x4*)&sA[cur][ar][ak + 16];
    af1.q[0] = *(const u32x4*)&sA[cur][ar + 16][ak];
    af1.q[1] = *(const u32x4*)&sA[cur][ar + 16][ak + 16];
#pragma unroll
    for (int j = 0; j < 8; ++j) {
      const int br = j * 16 + ll;
      Frag16 bfr;
      bfr.q[0] = *(const u32x4*)&sB[cur][br][bk];
      bfr.q[1] = *(const u32x4*)&sB[cur][br][bk + 16];
      acc[0][j] = __builtin_amdgcn_wmma_f32_16x16x32_bf16(
          false, af0.v, false, bfr.v, (short)0, acc[0][j], false, false);
      acc[1][j] = __builtin_amdgcn_wmma_f32_16x16x32_bf16(
          false, af1.v, false, bfr.v, (short)0, acc[1][j], false, false);
    }
    wait_async0();
    __syncthreads();
  }

  // C layout: VGPR r -> row r (lanes 0-15) / r+8 (lanes 16-31), col = lane&15
#pragma unroll
  for (int mi = 0; mi < 2; ++mi) {
    const int mr = m0 + wave * 32 + mi * 16 + (lh << 3);
#pragma unroll
    for (int j = 0; j < 8; ++j) {
      const int gn = n0 + j * 16 + ll;
      const float bn = bias[gn];
#pragma unroll
      for (int r = 0; r < 8; ++r) {
        float v = acc[mi][j][r] + bn;
        const size_t idx = (size_t)(mr + r) * Ndim + gn;
        if (clipToBf16) {
          v = fminf(fmaxf(v, -CLIP_), CLIP_);
          outB[idx] = f2bf(v);
        } else {
          outF[idx] = v;
        }
      }
    }
  }
}

// ------------------------------------------------------- LayerNorm + split ---
// blockIdx.x = b*S+s ; blockIdx.y: 0=q(LN), 1=k(LN), 2=v(transpose to [b,h,dh,s])
__global__ __launch_bounds__(256) void ln_split_kernel(
    const __bf16* __restrict__ qkv,
    const float* __restrict__ q_sc, const float* __restrict__ q_bi,
    const float* __restrict__ k_sc, const float* __restrict__ k_bi,
    __bf16* __restrict__ qh, __bf16* __restrict__ kh, __bf16* __restrict__ vt)
{
  const int row   = blockIdx.x;
  const int which = blockIdx.y;
  const int b = row / S_;
  const int s = row - b * S_;
  const __bf16* src = qkv + (size_t)row * N_QKV + which * D_;
  const int tid = threadIdx.x;

  if (which == 2) {
    for (int c = tid; c < D_; c += 256) {
      const int h = c >> 7, off = c & 127;
      vt[(((size_t)b * H_ + h) * DH_ + off) * S_ + s] = src[c];
    }
    return;
  }

  float vals[8];
  float sum = 0.f, sq = 0.f;
#pragma unroll
  for (int i = 0; i < 8; ++i) {
    const float x = bf2f(src[tid + i * 256]);
    vals[i] = x; sum += x; sq += x * x;
  }
#pragma unroll
  for (int off = 1; off < 32; off <<= 1) {
    sum += __shfl_xor(sum, off, 32);
    sq  += __shfl_xor(sq,  off, 32);
  }
  __shared__ float wsum[8], wsq[8];
  if ((tid & 31) == 0) { wsum[tid >> 5] = sum; wsq[tid >> 5] = sq; }
  __syncthreads();
  float ts = 0.f, tq = 0.f;
#pragma unroll
  for (int w = 0; w < 8; ++w) { ts += wsum[w]; tq += wsq[w]; }
  const float mean = ts * (1.0f / (float)D_);
  const float var  = tq * (1.0f / (float)D_) - mean * mean;
  const float rstd = rsqrtf(var + 1e-5f);
  const float* sc = which ? k_sc : q_sc;
  const float* bi = which ? k_bi : q_bi;
  __bf16* dst = which ? kh : qh;
#pragma unroll
  for (int i = 0; i < 8; ++i) {
    const int c = tid + i * 256;
    const float y = (vals[i] - mean) * rstd * sc[c] + bi[c];
    const int h = c >> 7, off = c & 127;
    dst[(((size_t)b * H_ + h) * S_ + s) * DH_ + off] = f2bf(y);
  }
}

// -------------------------------------------------------- flash attention ---
// grid: (b*H, S/128). Block: 8 waves, each owns 16 q rows, full dh=128.
__global__ __launch_bounds__(256) void attn_kernel(
    const __bf16* __restrict__ qh, const __bf16* __restrict__ kh,
    const __bf16* __restrict__ vt, __bf16* __restrict__ ctxb)
{
  __shared__ __align__(16) __bf16 sK[32][136];     // K rows [kcol][dh]
  __shared__ __align__(16) __bf16 sV[128][40];     // V^T [dh][kcol]
  __shared__ __align__(16) __bf16 sP[8][16][40];   // per-wave P tile
  const int bh = blockIdx.x;            // b*H + h
  const int qb = blockIdx.y;            // q block of 128 rows
  const int b = bh >> 4, h = bh & 15;
  const int tid  = threadIdx.x;
  const int wave = tid >> 5;
  const int lane = tid & 31;
  const int lh = lane >> 4;
  const int ll = lane & 15;
  const int q0 = qb * 128 + wave * 16;

  const __bf16* Q = qh + (size_t)bh * S_ * DH_;
  const __bf16* K = kh + (size_t)bh * S_ * DH_;
  const __bf16* V = vt + (size_t)bh * DH_ * S_;

  // Q fragments resident: 4 k-steps of 32 covering dh=128
  Frag16 qf[4];
  {
    const int qr = q0 + ll;
#pragma unroll
    for (int f = 0; f < 4; ++f) {
      const int kb = f * 32 + lh * 8;
      qf[f].q[0] = *(const u32x4*)(Q + (size_t)qr * DH_ + kb);
      qf[f].q[1] = *(const u32x4*)(Q + (size_t)qr * DH_ + kb + 16);
    }
  }

  v8f o[8] = {};
  float m_run[8], l_run[8];
#pragma unroll
  for (int r = 0; r < 8; ++r) { m_run[r] = -3.0e38f; l_run[r] = 0.f; }

  const int nkb = (qb + 1) * 4;          // causal: k <= end of this q block
  for (int kb = 0; kb < nkb; ++kb) {
    const int kbase = kb * 32;
    __syncthreads();
    {
      const int r = tid >> 3, c = (tid & 7) * 16;
      const u32x4* g = (const u32x4*)(K + (size_t)(kbase + r) * DH_ + c);
      *(u32x4*)&sK[r][c]     = g[0];
      *(u32x4*)&sK[r][c + 8] = g[1];
    }
    {
      const int r = tid >> 1, c = (tid & 1) * 16;
      const u32x4* g = (const u32x4*)(V + (size_t)r * S_ + kbase + c);
      *(u32x4*)&sV[r][c]     = g[0];
      *(u32x4*)&sV[r][c + 8] = g[1];
    }
    __syncthreads();

    // S = Q K^T  (two 16-col tiles)
    v8f sc[2] = {};
    const int bk = lh * 8;
#pragma unroll
    for (int j = 0; j < 2; ++j) {
#pragma unroll
      for (int f = 0; f < 4; ++f) {
        Frag16 bfr;
        const int br = j * 16 + ll;
        bfr.q[0] = *(const u32x4*)&sK[br][f * 32 + bk];
        bfr.q[1] = *(const u32x4*)&sK[br][f * 32 + bk + 16];
        sc[j] = __builtin_amdgcn_wmma_f32_16x16x32_bf16(
            false, qf[f].v, false, bfr.v, (short)0, sc[j], false, false);
      }
    }

    // online softmax; a row's 16 cols live across one 16-lane half
#pragma unroll
    for (int r = 0; r < 8; ++r) {
      const int qr = q0 + r + (lh << 3);
      float s0 = sc[0][r] * SMAX_SCALE;
      float s1 = sc[1][r] * SMAX_SCALE;
      if (kbase + ll      > qr) s0 = -3.0e38f;
      if (kbase + 16 + ll > qr) s1 = -3.0e38f;
      float mt = fmaxf(s0, s1);
#pragma unroll
      for (int off = 1; off < 16; off <<= 1) mt = fmaxf(mt, __shfl_xor(mt, off, 16));
      const float mnew = fmaxf(m_run[r], mt);
      const float corr = __expf(m_run[r] - mnew);
      m_run[r] = mnew;
      const float e0 = __expf(s0 - mnew);
      const float e1 = __expf(s1 - mnew);
      float ps = e0 + e1;
#pragma unroll
      for (int off = 1; off < 16; off <<= 1) ps += __shfl_xor(ps, off, 16);
      l_run[r] = l_run[r] * corr + ps;
#pragma unroll
      for (int dj = 0; dj < 8; ++dj) o[dj][r] *= corr;
      sP[wave][r + (lh << 3)][ll]      = f2bf(e0);
      sP[wave][r + (lh << 3)][16 + ll] = f2bf(e1);
    }
    asm volatile("s_wait_dscnt 0" ::: "memory");   // same-wave P store->load

    // O += P V  (A frag of P from LDS, B = V^T rows)
    Frag16 pf;
    pf.q[0] = *(const u32x4*)&sP[wave][ll][lh * 8];
    pf.q[1] = *(const u32x4*)&sP[wave][ll][lh * 8 + 16];
#pragma unroll
    for (int dj = 0; dj < 8; ++dj) {
      Frag16 bfr;
      const int br = dj * 16 + ll;
      bfr.q[0] = *(const u32x4*)&sV[br][bk];
      bfr.q[1] = *(const u32x4*)&sV[br][bk + 16];
      o[dj] = __builtin_amdgcn_wmma_f32_16x16x32_bf16(
          false, pf.v, false, bfr.v, (short)0, o[dj], false, false);
    }
  }

  // normalize + write ctx as bf16 in [b, s, d] with d = h*128 + col
#pragma unroll
  for (int r = 0; r < 8; ++r) {
    const float inv = 1.0f / l_run[r];
    const int qr = q0 + r + (lh << 3);
#pragma unroll
    for (int dj = 0; dj < 8; ++dj) {
      const int dc = h * DH_ + dj * 16 + ll;
      ctxb[((size_t)b * S_ + qr) * D_ + dc] = f2bf(o[dj][r] * inv);
    }
  }
}

// ------------------------------------------------------------------ launch ---
extern "C" void kernel_launch(void* const* d_in, const int* in_sizes, int n_in,
                              void* d_out, int out_size, void* d_ws, size_t ws_size,
                              hipStream_t stream) {
  const float* x      = (const float*)d_in[0];
  const float* Wqkv_w = (const float*)d_in[1];
  const float* Wqkv_b = (const float*)d_in[2];
  const float* q_sc   = (const float*)d_in[3];
  const float* q_bi   = (const float*)d_in[4];
  const float* k_sc   = (const float*)d_in[5];
  const float* k_bi   = (const float*)d_in[6];
  const float* out_w  = (const float*)d_in[7];
  const float* out_b  = (const float*)d_in[8];
  float* out = (float*)d_out;

  char* ws = (char*)d_ws;
  size_t off = 0;
  auto alloc = [&](size_t bytes) {
    char* p = ws + off; off += (bytes + 255) & ~(size_t)255; return p;
  };
  __bf16* xb    = (__bf16*)alloc((size_t)M_ * D_ * 2);       // x bf16
  __bf16* wqkvb = (__bf16*)alloc((size_t)N_QKV * D_ * 2);    // Wqkv bf16
  __bf16* owb   = (__bf16*)alloc((size_t)D_ * D_ * 2);       // out_w bf16
  __bf16* qkvb  = (__bf16*)alloc((size_t)M_ * N_QKV * 2);    // clipped qkv
  __bf16* qhb   = (__bf16*)alloc((size_t)M_ * D_ * 2);       // q [b,h,s,dh]
  __bf16* khb   = (__bf16*)alloc((size_t)M_ * D_ * 2);       // k [b,h,s,dh]
  __bf16* vtb   = (__bf16*)alloc((size_t)M_ * D_ * 2);       // v^T [b,h,dh,s]
  __bf16* ctxb  = (__bf16*)alloc((size_t)M_ * D_ * 2);       // ctx [b,s,d]

  cvt_f32_bf16<<<4096, 256, 0, stream>>>(x,      xb,    M_ * D_);
  cvt_f32_bf16<<<4096, 256, 0, stream>>>(Wqkv_w, wqkvb, N_QKV * D_);
  cvt_f32_bf16<<<4096, 256, 0, stream>>>(out_w,  owb,   D_ * D_);

  gemm_bf16_wmma<<<dim3(M_ / 256, N_QKV / 128), 256, 0, stream>>>(
      xb, wqkvb, Wqkv_b, qkvb, nullptr, M_, N_QKV, D_, 1);

  ln_split_kernel<<<dim3(M_, 3), 256, 0, stream>>>(
      qkvb, q_sc, q_bi, k_sc, k_bi, qhb, khb, vtb);

  attn_kernel<<<dim3(B_ * H_, S_ / 128), 256, 0, stream>>>(qhb, khb, vtb, ctxb);

  gemm_bf16_wmma<<<dim3(M_ / 256, D_ / 128), 256, 0, stream>>>(
      ctxb, owb, out_b, nullptr, out, M_, D_, D_, 0);
}
